// GraphTransformerLayer_33457795236066
// MI455X (gfx1250) — matrix operand order repaired and checked by reading, not compiled, not run on hardware
//
#include <hip/hip_runtime.h>

#define D 128
#define H 4

typedef __attribute__((ext_vector_type(16))) _Float16 v16h;
typedef __attribute__((ext_vector_type(8)))  _Float16 v8h;
typedef __attribute__((ext_vector_type(4)))  _Float16 v4h;
typedef __attribute__((ext_vector_type(8)))  float    v8f;
typedef __attribute__((ext_vector_type(4)))  float    v4f;

// ---------------------------------------------------------------------------
// A-fragment builder for v_wmma_f32_16x16x32_f16.
// Contiguous k order used by this code, not the ISA table order (16-bit A, 16x32): lane L holds row M=L%16;
// lanes 0-15 hold K k0..k0+15; lanes 16-31 hold K k0+16..k0+31 (matches b_frag).
// `row` points at a row-major f16 row in LDS (16B-aligned strides).
// ---------------------------------------------------------------------------
__device__ __forceinline__ v16h a_frag(const _Float16* row, int k0, int hi) {
  v8h lo = *(const v8h*)(row + k0 + hi * 16);
  v8h hp = *(const v8h*)(row + k0 + hi * 16 + 8);
  v16h r;
#pragma unroll
  for (int i = 0; i < 8; ++i) { r[i] = lo[i]; r[i + 8] = hp[i]; }
  return r;
}

// ---------------------------------------------------------------------------
// B-fragment: weights pre-packed transposed: bt[col][K] f16 row-major.
// Lane L<16 holds col L, K=k0..k0+15; lane>=16 holds col L-16, K=k0+16..k0+31
// -> one contiguous 32B load per lane (2x global_load_b128, L2-resident).
// ---------------------------------------------------------------------------
__device__ __forceinline__ v16h b_frag(const _Float16* __restrict__ bt,
                                       int col, int Ks, int k0, int hi) {
  return *(const v16h*)(bt + (size_t)col * Ks + k0 + hi * 16);
}

// ---------------------------------------------------------------------------
__global__ void zero_kernel(float* __restrict__ p, int n) {
  int i = blockIdx.x * 256 + threadIdx.x;
  if (i < n) p[i] = 0.0f;
}

// Pack all weight matrices into f16, transposed to [out_col][K] layout.
__global__ void pack_kernel(const float* __restrict__ Wself,
                            const float* __restrict__ Wneigh,
                            const float* __restrict__ Wfc,
                            const float* __restrict__ W1,
                            const float* __restrict__ W2,
                            _Float16* __restrict__ bt1,   // [H][128 cols][256 K]
                            _Float16* __restrict__ btfc,  // [128 cols][512 K]
                            _Float16* __restrict__ btw1,  // [128 cols][128 K]
                            _Float16* __restrict__ btw2) {
  const int n1 = H * 128 * 256, n2 = 512 * 128, n3 = 128 * 128;
  int i = blockIdx.x * 256 + threadIdx.x;
  if (i < n1) {
    int k = i & 255, f = (i >> 8) & 127, h = i >> 15;
    float w = (k < 128) ? Wself[((h * 128 + k) * 128) + f]
                        : Wneigh[((h * 128 + (k - 128)) * 128) + f];
    bt1[i] = (_Float16)w;
  } else if (i < n1 + n2) {
    int j = i - n1, k = j & 511, f = j >> 9;
    btfc[j] = (_Float16)Wfc[k * 128 + f];
  } else if (i < n1 + n2 + n3) {
    int j = i - n1 - n2, k = j & 127, f = j >> 7;
    btw1[j] = (_Float16)W1[k * 128 + f];
  } else if (i < n1 + n2 + 2 * n3) {
    int j = i - n1 - n2 - n3, k = j & 127, f = j >> 7;
    btw2[j] = (_Float16)W2[k * 128 + f];
  }
}

// LayerNorm: one wave32 per row (4 floats/lane), shuffle reduction.
__global__ __launch_bounds__(256) void ln_kernel(
    const float* __restrict__ x, const float* __restrict__ g,
    const float* __restrict__ b, float* __restrict__ y32,
    _Float16* __restrict__ y16, int n) {
  int wave = threadIdx.x >> 5, lane = threadIdx.x & 31;
  int row = blockIdx.x * 8 + wave;
  if (row >= n) return;
  const float* xr = x + (size_t)row * D;
  v4f v = *(const v4f*)(xr + lane * 4);
  float s = v[0] + v[1] + v[2] + v[3];
  float s2 = v[0] * v[0] + v[1] * v[1] + v[2] * v[2] + v[3] * v[3];
#pragma unroll
  for (int o = 16; o > 0; o >>= 1) {
    s += __shfl_xor(s, o);
    s2 += __shfl_xor(s2, o);
  }
  float mu = s * (1.0f / D);
  float var = s2 * (1.0f / D) - mu * mu;
  float rinv = rsqrtf(var + 1e-5f);
  v4f gv = *(const v4f*)(g + lane * 4);
  v4f bv = *(const v4f*)(b + lane * 4);
  v4f y;
  v4h yh;
#pragma unroll
  for (int j = 0; j < 4; ++j) {
    float t = (v[j] - mu) * rinv * gv[j] + bv[j];
    y[j] = t;
    yh[j] = (_Float16)t;
  }
  if (y32) *(v4f*)(y32 + (size_t)row * D + lane * 4) = y;
  *(v4h*)(y16 + (size_t)row * D + lane * 4) = yh;
}

// Edge scatter: one wave32 per edge, 4 floats/lane, f32 atomics (the HBM
// roofline term: ~410MB random traffic; atomics go straight to L2).
__global__ __launch_bounds__(256) void scatter_kernel(
    const float* __restrict__ xn, const int* __restrict__ src,
    const int* __restrict__ dst, float* __restrict__ nsum,
    float* __restrict__ deg, int e) {
  int wave = threadIdx.x >> 5, lane = threadIdx.x & 31;
  int idx = blockIdx.x * 8 + wave;
  if (idx >= e) return;
  int s = src[idx], d = dst[idx];
  v4f v = *(const v4f*)(xn + (size_t)s * D + lane * 4);
  float* o = nsum + (size_t)d * D + lane * 4;
  atomicAdd(o + 0, v[0]);
  atomicAdd(o + 1, v[1]);
  atomicAdd(o + 2, v[2]);
  atomicAdd(o + 3, v[3]);
  if (lane == 0) atomicAdd(deg + d, 1.0f);
}

// ---------------------------------------------------------------------------
// Fused GNN block, M=80 rows per block (5 subtiles of 16), 8 wave32s, one
// 16-col tile per wave. Each B fragment is reused for 5 WMMAs -> 5x less
// weight traffic from L2. neigh_mean is computed in-flight from nsum/deg
// during LDS staging (mean kernel fused away). Dynamic LDS (~124KB of the
// WGP's 320KB).
// Stage 1: heads[h] = [xn | nm](80x256) @ bt1[h] + b_self, relu -> cat f16.
// Stage 2: xres = x + cat(80x512) @ btfc + b_fc.
// ---------------------------------------------------------------------------
__global__ __launch_bounds__(256) void gnn_kernel(
    const _Float16* __restrict__ xnh, const float* __restrict__ nsum,
    const float* __restrict__ deg, const _Float16* __restrict__ bt1,
    const _Float16* __restrict__ btfc, const float* __restrict__ bself,
    const float* __restrict__ bfc, const float* __restrict__ x,
    float* __restrict__ xres, int n) {
  extern __shared__ _Float16 smem[];
  _Float16* sXn  = smem;               // [80][128]
  _Float16* sNm  = smem + 80 * 128;    // [80][128]
  _Float16* sCat = smem + 2 * 80 * 128;  // [80][520] (padded stride)
  const int CATS = 520;
  int lane = threadIdx.x & 31, wave = threadIdx.x >> 5;
  int rowBase = blockIdx.x * 80;
  if (rowBase + 80 > n) return;        // N is a multiple of 80

  // Cooperative staging: xn tile (f16 copy) + neigh_mean tile (nsum/deg).
#pragma unroll
  for (int st = 0; st < 5; ++st) {
    int t = threadIdx.x;
    int r = st * 16 + (t >> 4);        // local row 0..79
    int c = (t & 15) * 8;
    int gr = rowBase + r;
    *(v8h*)(sXn + r * D + c) = *(const v8h*)(xnh + (size_t)gr * D + c);
    float rd = 1.0f / fmaxf(deg[gr], 1.0f);
    v4f a0 = *(const v4f*)(nsum + (size_t)gr * D + c);
    v4f a1 = *(const v4f*)(nsum + (size_t)gr * D + c + 4);
    v8h nm;
#pragma unroll
    for (int j = 0; j < 4; ++j) {
      nm[j]     = (_Float16)(a0[j] * rd);
      nm[j + 4] = (_Float16)(a1[j] * rd);
    }
    *(v8h*)(sNm + r * D + c) = nm;
  }
  __syncthreads();

  int col = wave * 16 + (lane & 15);
  int hi = lane >> 4;

  // ---- Stage 1: 4 heads, K=256 (xn | neigh_mean), 5 row-subtiles ----
#pragma unroll 1
  for (int h = 0; h < H; ++h) {
    const _Float16* bth = bt1 + (size_t)h * 128 * 256;
    v8f acc[5];
#pragma unroll
    for (int st = 0; st < 5; ++st) acc[st] = (v8f){};
#pragma unroll
    for (int kk = 0; kk < 8; ++kk) {
      int k0 = kk * 32;
      v16h b = b_frag(bth, col, 256, k0, hi);
      const _Float16* aBase = (k0 < 128) ? sXn : sNm;
      int ka = k0 & 127;
#pragma unroll
      for (int st = 0; st < 5; ++st) {
        v16h a = a_frag(aBase + (st * 16 + (lane & 15)) * D, ka, hi);
        acc[st] = __builtin_amdgcn_wmma_f32_16x16x32_f16(
            false, a, false, b, (short)0, acc[st], false, false);
      }
    }
    float bias = bself[h * 128 + col];
    int catCol = h * 128 + col;
#pragma unroll
    for (int st = 0; st < 5; ++st) {
#pragma unroll
      for (int i = 0; i < 8; ++i) {
        float v = acc[st][i] + bias;
        v = v > 0.f ? v : 0.f;           // relu
        sCat[(st * 16 + hi * 8 + i) * CATS + catCol] = (_Float16)v;
      }
    }
  }
  __syncthreads();

  // ---- Stage 2: gnn_out = cat(80x512) @ btfc + b_fc + x ----
  v8f acc2[5];
#pragma unroll
  for (int st = 0; st < 5; ++st) acc2[st] = (v8f){};
#pragma unroll
  for (int kk = 0; kk < 16; ++kk) {
    int k0 = kk * 32;
    v16h b = b_frag(btfc, col, 512, k0, hi);
#pragma unroll
    for (int st = 0; st < 5; ++st) {
      v16h a = a_frag(sCat + (st * 16 + (lane & 15)) * CATS, k0, hi);
      acc2[st] = __builtin_amdgcn_wmma_f32_16x16x32_f16(
          false, a, false, b, (short)0, acc2[st], false, false);
    }
  }
  float bias = bfc[col];
#pragma unroll
  for (int st = 0; st < 5; ++st) {
#pragma unroll
    for (int i = 0; i < 8; ++i) {
      int r = rowBase + st * 16 + hi * 8 + i;
      xres[(size_t)r * D + col] = x[(size_t)r * D + col] + acc2[st][i] + bias;
    }
  }
}

// ---------------------------------------------------------------------------
// Fused FFN block, M=80: t = relu(hn @ W1 + b1); out = xres + t @ W2 + b2.
// ---------------------------------------------------------------------------
__global__ __launch_bounds__(256) void ffn_kernel(
    const _Float16* __restrict__ hnh, const _Float16* __restrict__ btw1,
    const _Float16* __restrict__ btw2, const float* __restrict__ b1,
    const float* __restrict__ b2, const float* __restrict__ xres,
    float* __restrict__ out, int n) {
  __shared__ _Float16 sH[80 * D];       // 20KB
  __shared__ _Float16 sT[80 * 136];     // 21.25KB (padded stride)
  const int TS = 136;
  int lane = threadIdx.x & 31, wave = threadIdx.x >> 5;
  int rowBase = blockIdx.x * 80;
  if (rowBase + 80 > n) return;
#pragma unroll
  for (int st = 0; st < 5; ++st) {
    int t = threadIdx.x;
    int r = st * 16 + (t >> 4);
    int c = (t & 15) * 8;
    *(v8h*)(sH + r * D + c) = *(const v8h*)(hnh + (size_t)(rowBase + r) * D + c);
  }
  __syncthreads();

  int col = wave * 16 + (lane & 15);
  int hi = lane >> 4;

  v8f acc[5];
#pragma unroll
  for (int st = 0; st < 5; ++st) acc[st] = (v8f){};
#pragma unroll
  for (int kk = 0; kk < 4; ++kk) {
    int k0 = kk * 32;
    v16h b = b_frag(btw1, col, 128, k0, hi);
#pragma unroll
    for (int st = 0; st < 5; ++st) {
      v16h a = a_frag(sH + (st * 16 + (lane & 15)) * D, k0, hi);
      acc[st] = __builtin_amdgcn_wmma_f32_16x16x32_f16(
          false, a, false, b, (short)0, acc[st], false, false);
    }
  }
  float bias1 = b1[col];
#pragma unroll
  for (int st = 0; st < 5; ++st) {
#pragma unroll
    for (int i = 0; i < 8; ++i) {
      float v = acc[st][i] + bias1;
      v = v > 0.f ? v : 0.f;
      sT[(st * 16 + hi * 8 + i) * TS + col] = (_Float16)v;
    }
  }
  __syncthreads();

  v8f acc2[5];
#pragma unroll
  for (int st = 0; st < 5; ++st) acc2[st] = (v8f){};
#pragma unroll
  for (int kk = 0; kk < 4; ++kk) {
    int k0 = kk * 32;
    v16h b = b_frag(btw2, col, 128, k0, hi);
#pragma unroll
    for (int st = 0; st < 5; ++st) {
      v16h a = a_frag(sT + (st * 16 + (lane & 15)) * TS, k0, hi);
      acc2[st] = __builtin_amdgcn_wmma_f32_16x16x32_f16(
          false, a, false, b, (short)0, acc2[st], false, false);
    }
  }
  float bias2 = b2[col];
#pragma unroll
  for (int st = 0; st < 5; ++st) {
#pragma unroll
    for (int i = 0; i < 8; ++i) {
      int r = rowBase + st * 16 + hi * 8 + i;
      out[(size_t)r * D + col] = xres[(size_t)r * D + col] + acc2[st][i] + bias2;
    }
  }
}

// ---------------------------------------------------------------------------
extern "C" void kernel_launch(void* const* d_in, const int* in_sizes, int n_in,
                              void* d_out, int out_size, void* d_ws,
                              size_t ws_size, hipStream_t stream) {
  const float* x      = (const float*)d_in[0];
  const int*   src    = (const int*)d_in[1];
  const int*   dst    = (const int*)d_in[2];
  const float* ln1g   = (const float*)d_in[3];
  const float* ln1b   = (const float*)d_in[4];
  const float* Wneigh = (const float*)d_in[5];
  const float* Wself  = (const float*)d_in[6];
  const float* bself  = (const float*)d_in[7];
  const float* Wfc    = (const float*)d_in[8];
  const float* bfc    = (const float*)d_in[9];
  const float* ln2g   = (const float*)d_in[10];
  const float* ln2b   = (const float*)d_in[11];
  const float* W1     = (const float*)d_in[12];
  const float* b1     = (const float*)d_in[13];
  const float* W2     = (const float*)d_in[14];
  const float* b2     = (const float*)d_in[15];
  float* out = (float*)d_out;

  const int N = in_sizes[0] / D;
  const int E = in_sizes[1];

  char* ws = (char*)d_ws;
  size_t off = 0;
  auto alloc = [&](size_t bytes) {
    size_t o = off;
    off += (bytes + 255) & ~(size_t)255;
    return o;
  };
  float*    xn32 = (float*)(ws + alloc((size_t)N * D * 4));
  _Float16* xnh  = (_Float16*)(ws + alloc((size_t)N * D * 2));
  float*    nsum = (float*)(ws + alloc((size_t)N * D * 4 + (size_t)N * 4));
  float*    deg  = nsum + (size_t)N * D;  // contiguous with nsum: one zero pass
  float*    xres = (float*)(ws + alloc((size_t)N * D * 4));
  _Float16* hnh  = (_Float16*)(ws + alloc((size_t)N * D * 2));
  _Float16* bt1  = (_Float16*)(ws + alloc((size_t)H * 128 * 256 * 2));
  _Float16* btfc = (_Float16*)(ws + alloc((size_t)512 * 128 * 2));
  _Float16* btw1 = (_Float16*)(ws + alloc((size_t)128 * 128 * 2));
  _Float16* btw2 = (_Float16*)(ws + alloc((size_t)128 * 128 * 2));

  const int nzero = N * D + N;
  zero_kernel<<<(nzero + 255) / 256, 256, 0, stream>>>(nsum, nzero);

  const int npack = H * 128 * 256 + 512 * 128 + 2 * 128 * 128;
  pack_kernel<<<(npack + 255) / 256, 256, 0, stream>>>(
      Wself, Wneigh, Wfc, W1, W2, bt1, btfc, btw1, btw2);

  ln_kernel<<<(N + 7) / 8, 256, 0, stream>>>(x, ln1g, ln1b, xn32, xnh, N);

  scatter_kernel<<<(E + 7) / 8, 256, 0, stream>>>(xn32, src, dst, nsum, deg, E);

  const size_t gnnLds = (size_t)(2 * 80 * 128 + 80 * 520) * sizeof(_Float16);
  gnn_kernel<<<(N + 79) / 80, 256, gnnLds, stream>>>(
      xnh, nsum, deg, bt1, btfc, bself, bfc, x, xres, N);

  ln_kernel<<<(N + 7) / 8, 256, 0, stream>>>(xres, ln2g, ln2b, nullptr, hnh, N);

  ffn_kernel<<<(N + 79) / 80, 256, 0, stream>>>(hnh, btw1, btw2, b1, b2, xres,
                                                out, N);
}
